// Regressor_56341380989616
// MI455X (gfx1250) — compile-verified, matrix operand order checked by reading, not run on hardware
//
#include <hip/hip_runtime.h>
#include <math.h>

// Problem constants (from reference)
#define B_    16384
#define E_    128
#define H_    1024
#define V_    32
#define KTOT  1280    // 2E + H (fused GEMM K)
#define GATES 4096    // 4H

typedef __bf16 bf16_t;
typedef __bf16 bf16x16 __attribute__((ext_vector_type(16)));
typedef float  f32x8   __attribute__((ext_vector_type(8)));

// Hardware-native transcendentals: v_exp_f32 + v_rcp_f32 instead of libm
// polynomial expansions (libm tanhf alone is ~10x the VALU ops).
__device__ __forceinline__ float fast_rcp(float x) { return __builtin_amdgcn_rcpf(x); }
__device__ __forceinline__ float sigf(float x) { return fast_rcp(1.0f + __expf(-x)); }
// tanh(x) = (e^{2x}-1)/(e^{2x}+1) = 1 - 2/(e^{2x}+1); saturates correctly at +/-inf.
__device__ __forceinline__ float tanh_fast(float x) {
    return 1.0f - 2.0f * fast_rcp(1.0f + __expf(2.0f * x));
}

// ---------------------------------------------------------------------------
// Kernel 1a: pack activation matrix Xcat[B,1280] (bf16):
//   cols [0,128)    = embedding[parent[b]]
//   cols [128,256)  = embedding[sibling[b]]
//   cols [256,1280) = h[b]
// ---------------------------------------------------------------------------
__global__ void pack_x(const int* __restrict__ parent, const int* __restrict__ sibling,
                       const float* __restrict__ h, const float* __restrict__ emb,
                       bf16_t* __restrict__ xcat) {
    int b = blockIdx.x;
    int p = parent[b];
    int s = sibling[b];
    bf16_t* row = xcat + (size_t)b * KTOT;
    for (int col = threadIdx.x; col < KTOT; col += blockDim.x) {
        float v;
        if (col < E_)            v = emb[p * E_ + col];
        else if (col < 2 * E_)   v = emb[s * E_ + (col - E_)];
        else                     v = h[(size_t)b * H_ + (col - 2 * E_)];
        row[col] = (bf16_t)v;
    }
}

// ---------------------------------------------------------------------------
// Kernel 1b: pack weight matrix Wb[4096,1280] (bf16) = [W_ih | W_hh] rows,
// and bias[4096] = b_ih + b_hh.
// ---------------------------------------------------------------------------
__global__ void pack_w(const float* __restrict__ W_ih, const float* __restrict__ W_hh,
                       const float* __restrict__ b_ih, const float* __restrict__ b_hh,
                       bf16_t* __restrict__ wb, float* __restrict__ bias) {
    int n = blockIdx.x;  // 0..4095
    bf16_t* row = wb + (size_t)n * KTOT;
    for (int col = threadIdx.x; col < KTOT; col += blockDim.x) {
        float v = (col < 2 * E_) ? W_ih[(size_t)n * (2 * E_) + col]
                                 : W_hh[(size_t)n * H_ + (col - 2 * E_)];
        row[col] = (bf16_t)v;
    }
    if (threadIdx.x == 0) bias[n] = b_ih[n] + b_hh[n];
}

// ---------------------------------------------------------------------------
// Kernel 2: fused gate GEMM (bf16 WMMA, f32 accum) + LSTM cell epilogue.
// gates[m,n] = sum_k Xcat[m,k] * Wb[n,k]   (n in {g*1024 + hcol})
// Wave tile: M=64, 16 H-columns, all 4 gates -> acc[4][4] (128 VGPRs).
// Block: 8 waves = 2(M) x 4(Hcol) -> 128 rows x 64 H-cols per block.
// ---------------------------------------------------------------------------
__global__ __launch_bounds__(256, 1)
void lstm_gemm(const bf16_t* __restrict__ xcat, const bf16_t* __restrict__ wb,
               const float* __restrict__ bias, const float* __restrict__ c_in,
               float* __restrict__ out_h, float* __restrict__ out_c) {
    const int lane  = threadIdx.x & 31;
    const int wave  = threadIdx.x >> 5;
    const int wm    = wave & 1;           // 0..1  (M direction)
    const int wn    = wave >> 1;          // 0..3  (H-col direction)
    const int mBase = blockIdx.x * 128 + wm * 64;
    const int nBase = blockIdx.y * 64 + wn * 16;

    // Fragment addressing per ISA 16-bit A/B layouts:
    // lane holds row/col (lane&15); lanes 0-15 hold K {0..7,16..23},
    // lanes 16-31 hold K {8..15,24..31} (two b128 chunks per fragment).
    const int lr   = lane & 15;
    const int koff = (lane >> 4) * 8;

    const bf16_t* aptr[4];
#pragma unroll
    for (int mt = 0; mt < 4; ++mt)
        aptr[mt] = xcat + (size_t)(mBase + mt * 16 + lr) * KTOT + koff;

    const bf16_t* bptr[4];
#pragma unroll
    for (int g = 0; g < 4; ++g)
        bptr[g] = wb + (size_t)(g * H_ + nBase + lr) * KTOT + koff;

    f32x8 zero;
#pragma unroll
    for (int j = 0; j < 8; ++j) zero[j] = 0.0f;

    f32x8 acc[4][4];  // [gate][m-tile]
#pragma unroll
    for (int g = 0; g < 4; ++g)
#pragma unroll
        for (int mt = 0; mt < 4; ++mt) acc[g][mt] = zero;

    union Frag { bf16x16 v; uint4 u[2]; };

    for (int k = 0; k < KTOT; k += 32) {
        Frag a[4], b[4];
#pragma unroll
        for (int mt = 0; mt < 4; ++mt) {
            a[mt].u[0] = *reinterpret_cast<const uint4*>(aptr[mt] + k);
            a[mt].u[1] = *reinterpret_cast<const uint4*>(aptr[mt] + k + 16);
        }
#pragma unroll
        for (int g = 0; g < 4; ++g) {
            b[g].u[0] = *reinterpret_cast<const uint4*>(bptr[g] + k);
            b[g].u[1] = *reinterpret_cast<const uint4*>(bptr[g] + k + 16);
        }
#pragma unroll
        for (int g = 0; g < 4; ++g)
#pragma unroll
            for (int mt = 0; mt < 4; ++mt)
                acc[g][mt] = __builtin_amdgcn_wmma_f32_16x16x32_bf16(
                    false, a[mt].v, false, b[g].v, (short)0, acc[g][mt], false, false);
    }

    // Epilogue: C/D layout -> VGPR j, lane l: row = j + 8*(l>>4), col = l&15.
    const int col = nBase + lr;
    const float bi = bias[0 * H_ + col];
    const float bf = bias[1 * H_ + col];
    const float bg = bias[2 * H_ + col];
    const float bo = bias[3 * H_ + col];
    const int rowadd = (lane >> 4) * 8;

#pragma unroll
    for (int mt = 0; mt < 4; ++mt) {
#pragma unroll
        for (int j = 0; j < 8; ++j) {
            const int row = mBase + mt * 16 + rowadd + j;
            const size_t idx = (size_t)row * H_ + col;
            const float gi = acc[0][mt][j] + bi;
            const float gf = acc[1][mt][j] + bf;
            const float gg = acc[2][mt][j] + bg;
            const float go = acc[3][mt][j] + bo;
            const float cv = c_in[idx];
            const float cn = sigf(gf) * cv + sigf(gi) * tanh_fast(gg);
            const float hn = sigf(go) * tanh_fast(cn);
            out_c[idx] = cn;
            out_h[idx] = hn;
        }
    }
}

// ---------------------------------------------------------------------------
// Kernel 3: logits = c_new @ W_fc.T + b_fc, then softmax. V==32==wave32:
// lane v owns logit v; W_fc (128KB) stays cache-resident across all rows.
// ---------------------------------------------------------------------------
__global__ __launch_bounds__(256)
void fc_softmax(const float* __restrict__ c_new, const float* __restrict__ W_fc,
                const float* __restrict__ b_fc, float* __restrict__ prob) {
    const int lane = threadIdx.x & 31;
    const int wave = threadIdx.x >> 5;
    const int b = blockIdx.x * 8 + wave;

    const float* crow = c_new + (size_t)b * H_;
    const float* wrow = W_fc + (size_t)lane * H_;

    float acc = b_fc[lane];
    for (int hh = 0; hh < H_; hh += 4) {
        float4 w  = *reinterpret_cast<const float4*>(wrow + hh);
        float4 cv = *reinterpret_cast<const float4*>(crow + hh);
        acc += w.x * cv.x + w.y * cv.y + w.z * cv.z + w.w * cv.w;
    }

    float m = acc;
#pragma unroll
    for (int off = 16; off > 0; off >>= 1)
        m = fmaxf(m, __shfl_xor(m, off, 32));
    const float e = __expf(acc - m);
    float s = e;
#pragma unroll
    for (int off = 16; off > 0; off >>= 1)
        s += __shfl_xor(s, off, 32);

    prob[(size_t)b * V_ + lane] = e * fast_rcp(s);
}

// ---------------------------------------------------------------------------
extern "C" void kernel_launch(void* const* d_in, const int* in_sizes, int n_in,
                              void* d_out, int out_size, void* d_ws, size_t ws_size,
                              hipStream_t stream) {
    const int*   parent  = (const int*)d_in[0];
    const int*   sibling = (const int*)d_in[1];
    const float* h       = (const float*)d_in[2];
    const float* c       = (const float*)d_in[3];
    const float* emb     = (const float*)d_in[4];
    const float* W_ih    = (const float*)d_in[5];
    const float* W_hh    = (const float*)d_in[6];
    const float* b_ih    = (const float*)d_in[7];
    const float* b_hh    = (const float*)d_in[8];
    const float* W_fc    = (const float*)d_in[9];
    const float* b_fc    = (const float*)d_in[10];

    // Workspace carve-up (~52.4 MB)
    char* ws = (char*)d_ws;
    bf16_t* xcat = (bf16_t*)ws;                                       // B*1280 bf16
    bf16_t* wb   = (bf16_t*)(ws + (size_t)B_ * KTOT * 2);             // 4096*1280 bf16
    float*  bias = (float*)(ws + (size_t)B_ * KTOT * 2
                               + (size_t)GATES * KTOT * 2);           // 4096 f32

    // Output layout: prob [B,V], h_new [B,H], c_new [B,H] (flat, in order)
    float* out_prob = (float*)d_out;
    float* out_h    = out_prob + (size_t)B_ * V_;
    float* out_c    = out_h + (size_t)B_ * H_;

    pack_x<<<B_, 256, 0, stream>>>(parent, sibling, h, emb, xcat);
    pack_w<<<GATES, 256, 0, stream>>>(W_ih, W_hh, b_ih, b_hh, wb, bias);

    dim3 grid(B_ / 128, H_ / 64);  // 128 x 16 blocks
    lstm_gemm<<<grid, 256, 0, stream>>>(xcat, wb, bias, c, out_h, out_c);

    fc_softmax<<<B_ / 8, 256, 0, stream>>>(out_c, W_fc, b_fc, out_prob);
}
